// Transformer_82429012345285
// MI455X (gfx1250) — matrix operand order repaired and checked by reading, not compiled, not run on hardware
//
#include <hip/hip_runtime.h>
#include <hip/hip_bf16.h>

// ---------------------------------------------------------------------------
// Transformer block for MI455X (gfx1250, wave32, WMMA).
//   B=2, S=2048, C=U=1024, H=16, dh=64, FFH=4096
// ---------------------------------------------------------------------------

typedef __attribute__((ext_vector_type(16))) _Float16 v16h;
typedef __attribute__((ext_vector_type(8)))  _Float16 v8h;
typedef __attribute__((ext_vector_type(8)))  float    v8f;

constexpr int BATCH  = 2;
constexpr int SEQ    = 2048;
constexpr int UDIM   = 1024;
constexpr int DH     = 64;
constexpr int FFH    = 4096;
constexpr int MROWS  = BATCH * SEQ;            // 4096
constexpr float NEGF = -1e9f;
constexpr float EPSF = 1e-5f;

union V16U { v16h v; v8h h[2]; };

__device__ __forceinline__ float4 ld4(const float* p) {
  return *(const float4*)p;
}

// Pack 16 f32 (two contiguous 8-float runs already gathered) into a v16h frag.
__device__ __forceinline__ v16h cvt16(float4 a0, float4 a1, float4 a2, float4 a3) {
  v16h r;
  r[0]  = (_Float16)a0.x; r[1]  = (_Float16)a0.y; r[2]  = (_Float16)a0.z; r[3]  = (_Float16)a0.w;
  r[4]  = (_Float16)a1.x; r[5]  = (_Float16)a1.y; r[6]  = (_Float16)a1.z; r[7]  = (_Float16)a1.w;
  r[8]  = (_Float16)a2.x; r[9]  = (_Float16)a2.y; r[10] = (_Float16)a2.z; r[11] = (_Float16)a2.w;
  r[12] = (_Float16)a3.x; r[13] = (_Float16)a3.y; r[14] = (_Float16)a3.z; r[15] = (_Float16)a3.w;
  return r;
}
__device__ __forceinline__ v16h cvt16s(float s, float4 a0, float4 a1, float4 a2, float4 a3) {
  v16h r;
  r[0]  = (_Float16)(s*a0.x); r[1]  = (_Float16)(s*a0.y); r[2]  = (_Float16)(s*a0.z); r[3]  = (_Float16)(s*a0.w);
  r[4]  = (_Float16)(s*a1.x); r[5]  = (_Float16)(s*a1.y); r[6]  = (_Float16)(s*a1.z); r[7]  = (_Float16)(s*a1.w);
  r[8]  = (_Float16)(s*a2.x); r[9]  = (_Float16)(s*a2.y); r[10] = (_Float16)(s*a2.z); r[11] = (_Float16)(s*a2.w);
  r[12] = (_Float16)(s*a3.x); r[13] = (_Float16)(s*a3.y); r[14] = (_Float16)(s*a3.z); r[15] = (_Float16)(s*a3.w);
  return r;
}

__device__ __forceinline__ v8f wmma32(v16h a, v16h b, v8f c) {
  // v_wmma_f32_16x16x32_f16 : D = A(16x32) x B(32x16) + C
  return __builtin_amdgcn_wmma_f32_16x16x32_f16(false, a, false, b, (short)0, c,
                                                false, false);
}

// ---------------------------------------------------------------------------
// GEMM:  C[M,N] = act( A[M,K] @ W[K,N] + bias ) (+ resid)
// block = 256 threads = 8 waves; macro tile 128(M) x 64(N).
// W-tile (32x64) staged in LDS as f16 PRE-SWIZZLED into B-fragment layout:
//   Bl[(t*32 + lane)*16 + e]  holds W[k = (lane>>4)*16 + e][n = t*16 + (lane&15)]
// so each wave reads a whole v16h per fragment (2x ds_load_b128).
// ---------------------------------------------------------------------------
__global__ __launch_bounds__(256) void wmma_gemm_kernel(
    const float* __restrict__ A, const float* __restrict__ W,
    const float* __restrict__ bias, const float* __restrict__ resid,
    float* __restrict__ Cmat, int M, int K, int N, int relu_flag)
{
  __shared__ __align__(32) _Float16 Bl[4 * 32 * 16];   // 4 KB

  const int tid   = threadIdx.x;
  const int lane  = tid & 31;
  const int wave  = tid >> 5;
  const int nBlkN = N >> 6;
  const int bm    = (int)blockIdx.x / nBlkN;
  const int bn    = (int)blockIdx.x % nBlkN;
  const int m0    = bm * 128 + wave * 16;
  const int n0    = bn * 64;
  const int rowA  = m0 + (lane & 15);
  const int hi8   = (lane >> 4) << 3;
  const int colc  = lane & 15;

  v8f acc[4] = {};

  // staging coords for this thread: 8 contiguous floats of the W tile
  const int si  = tid * 8;            // 0..2047
  const int skk = si >> 6;            // K row within tile
  const int snn = si & 63;            // N col base (multiple of 8)

  for (int k0 = 0; k0 < K; k0 += 32) {
    // ---- cooperative stage: W[k0:k0+32, n0:n0+64] -> swizzled f16 LDS ----
    {
      const float* wp = W + (k0 + skk) * N + n0 + snn;
      float4 w0 = ld4(wp), w1 = ld4(wp + 4);
      float wv_[8] = {w0.x, w0.y, w0.z, w0.w, w1.x, w1.y, w1.z, w1.w};
#pragma unroll
      for (int j = 0; j < 8; ++j) {
        const int nn   = snn + j;
        const int t    = nn >> 4;
        const int slot = ((skk >> 4) << 4) + (nn & 15);
        Bl[((t * 32 + slot) << 4) + (skk & 15)] = (_Float16)wv_[j];
      }
    }
    __syncthreads();

    // ---- A fragment: one contiguous 16-float run per lane ----
    const float* Ap = A + rowA * K + k0 + ((lane >> 4) << 4);
    __builtin_prefetch(Ap + 32, 0, 1);        // prefetch next K-chunk
    v16h a = cvt16(ld4(Ap), ld4(Ap + 4), ld4(Ap + 8), ld4(Ap + 12));

#pragma unroll
    for (int t = 0; t < 4; ++t) {
      v16h b = *(const v16h*)&Bl[(t * 32 + lane) << 4];   // 32B contiguous
      acc[t] = wmma32(a, b, acc[t]);
    }
    __syncthreads();
  }

#pragma unroll
  for (int t = 0; t < 4; ++t) {
    const int col = n0 + t * 16 + colc;
    const float bv = bias ? bias[col] : 0.f;
#pragma unroll
    for (int r = 0; r < 8; ++r) {
      const int row = m0 + r + hi8;
      float v = acc[t][r] + bv;
      if (resid) v += resid[row * N + col];
      if (relu_flag) v = fmaxf(v, 0.f);
      Cmat[row * N + col] = v;
    }
  }
}

// ---------------------------------------------------------------------------
// Fused causal flash attention. One wave = one (batch, head, 16-query tile).
// 32 keys per iteration: 4 score WMMAs + online softmax + 4 P*V WMMAs (K=32).
// Writes softmax(QK^T/sqrt(dh)) @ V  + x  (residual) into Out.
// ---------------------------------------------------------------------------
__global__ __launch_bounds__(256) void attn_kernel(
    const float* __restrict__ Q, const float* __restrict__ Kbuf,
    const float* __restrict__ Vbuf, const float* __restrict__ X,
    float* __restrict__ Out)
{
  __shared__ __align__(32) _Float16 Pl[8 * 16 * 32];   // per-wave 16x32 P tile

  const int lane = threadIdx.x & 31;
  const int wave = threadIdx.x >> 5;
  const int wid  = (int)blockIdx.x * 8 + wave;
  const int qt   = wid & 127;                 // SEQ/16 = 128 query tiles
  const int h    = (wid >> 7) & 15;
  const int b    = wid >> 11;
  const int q0   = qt << 4;
  const int hcol = h * DH;
  const int colk = lane & 15;
  const int hi8  = (lane >> 4) << 3;
  const int kb16 = (lane >> 4) << 4;
  const int rowbase = b * SEQ + q0;
  _Float16* Pw = Pl + wave * 512;

  // Q fragments, pre-scaled by 1/sqrt(dh) = 0.125
  const float* qrow = Q + (rowbase + colk) * UDIM + hcol + kb16;
  const v16h aQ0 = cvt16s(0.125f, ld4(qrow),      ld4(qrow + 4),
                                  ld4(qrow + 8),  ld4(qrow + 12));
  const v16h aQ1 = cvt16s(0.125f, ld4(qrow + 32), ld4(qrow + 36),
                                  ld4(qrow + 40), ld4(qrow + 44));

  float mrow[8], lrow[8];
#pragma unroll
  for (int r = 0; r < 8; ++r) { mrow[r] = -3.0e38f; lrow[r] = 0.f; }
  v8f acc[4] = {};

  for (int j0 = 0; j0 <= q0; j0 += 32) {
    // B-fragments of K^T : n = key (lane&15), reduction k = dh ((lane>>4)*16+e)
    const float* kr0 = Kbuf + (b * SEQ + j0 +      colk) * UDIM + hcol + kb16;
    const float* kr1 = Kbuf + (b * SEQ + j0 + 16 + colk) * UDIM + hcol + kb16;
    const v16h bK00 = cvt16(ld4(kr0),      ld4(kr0 + 4),  ld4(kr0 + 8),  ld4(kr0 + 12));
    const v16h bK01 = cvt16(ld4(kr0 + 32), ld4(kr0 + 36), ld4(kr0 + 40), ld4(kr0 + 44));
    const v16h bK10 = cvt16(ld4(kr1),      ld4(kr1 + 4),  ld4(kr1 + 8),  ld4(kr1 + 12));
    const v16h bK11 = cvt16(ld4(kr1 + 32), ld4(kr1 + 36), ld4(kr1 + 40), ld4(kr1 + 44));

    v8f s0 = {}, s1 = {};
    s0 = wmma32(aQ0, bK00, s0);  s0 = wmma32(aQ1, bK01, s0);
    s1 = wmma32(aQ0, bK10, s1);  s1 = wmma32(aQ1, bK11, s1);

    // causal mask + online softmax (row = r + hi8, col = colk)
#pragma unroll
    for (int r = 0; r < 8; ++r) {
      const int qrow_i = q0 + r + hi8;
      if (j0 + colk      > qrow_i) s0[r] = NEGF;
      if (j0 + 16 + colk > qrow_i) s1[r] = NEGF;
      float mx = fmaxf(s0[r], s1[r]);
      mx = fmaxf(mx, __shfl_xor(mx, 1, 32));
      mx = fmaxf(mx, __shfl_xor(mx, 2, 32));
      mx = fmaxf(mx, __shfl_xor(mx, 4, 32));
      mx = fmaxf(mx, __shfl_xor(mx, 8, 32));
      const float mnew = fmaxf(mrow[r], mx);
      const float corr = __expf(mrow[r] - mnew);
      const float p0 = __expf(s0[r] - mnew);
      const float p1 = __expf(s1[r] - mnew);
      float rs = p0 + p1;
      rs += __shfl_xor(rs, 1, 32);
      rs += __shfl_xor(rs, 2, 32);
      rs += __shfl_xor(rs, 4, 32);
      rs += __shfl_xor(rs, 8, 32);
      lrow[r] = lrow[r] * corr + rs;
      mrow[r] = mnew;
#pragma unroll
      for (int t = 0; t < 4; ++t) acc[t][r] *= corr;
      Pw[(r + hi8) * 32 +      colk] = (_Float16)p0;
      Pw[(r + hi8) * 32 + 16 + colk] = (_Float16)p1;
    }
    __builtin_amdgcn_wave_barrier();

    // P tile: C-layout -> A-layout via LDS; per lane two contiguous v8h runs
    V16U ap;
    ap.h[0] = *(const v8h*)&Pw[colk * 32 + kb16];
    ap.h[1] = *(const v8h*)&Pw[colk * 32 + kb16 + 8];

    // P @ V  (reduction k = key, n = dh column); lanes coalesce along n
#pragma unroll
    for (int t = 0; t < 4; ++t) {
      v16h bV;
      const float* vrow = Vbuf + (b * SEQ + j0 + kb16) * UDIM + hcol + t * 16 + colk;
#pragma unroll
      for (int e = 0; e < 16; ++e) bV[e] = (_Float16)vrow[e * UDIM];
      acc[t] = wmma32(ap.v, bV, acc[t]);
    }
    __builtin_amdgcn_wave_barrier();
  }

  // normalize + residual add
#pragma unroll
  for (int t = 0; t < 4; ++t) {
#pragma unroll
    for (int r = 0; r < 8; ++r) {
      const int row = rowbase + r + hi8;
      const int col = hcol + t * 16 + colk;
      Out[row * UDIM + col] = acc[t][r] / lrow[r] + X[row * UDIM + col];
    }
  }
}

// ---------------------------------------------------------------------------
// LayerNorm over (S,U) per batch — two pass.
// ---------------------------------------------------------------------------
__global__ void zero_kernel(float* p, int n) {
  int i = blockIdx.x * blockDim.x + threadIdx.x;
  if (i < n) p[i] = 0.f;
}

__global__ __launch_bounds__(256) void ln_reduce_kernel(
    const float* __restrict__ x, float* __restrict__ sums, int per_batch)
{
  __shared__ float rs[256], rq[256];
  const int b = blockIdx.y;
  const float* xb = x + (size_t)b * per_batch;
  float s = 0.f, sq = 0.f;
  for (int i = blockIdx.x * blockDim.x + threadIdx.x; i < per_batch;
       i += blockDim.x * gridDim.x) {
    const float v = xb[i];
    s += v; sq += v * v;
  }
  const int tid = threadIdx.x;
  rs[tid] = s; rq[tid] = sq;
  __syncthreads();
  for (int w = 128; w > 0; w >>= 1) {
    if (tid < w) { rs[tid] += rs[tid + w]; rq[tid] += rq[tid + w]; }
    __syncthreads();
  }
  if (tid == 0) {
    atomicAdd(&sums[b * 2 + 0], rs[0]);
    atomicAdd(&sums[b * 2 + 1], rq[0]);
  }
}

__global__ __launch_bounds__(256) void ln_apply_kernel(
    const float* __restrict__ x, const float* __restrict__ sums,
    const float* __restrict__ g, const float* __restrict__ bb,
    float* __restrict__ y, int per_batch)
{
  const int b = blockIdx.y;
  const float* xb = x + (size_t)b * per_batch;
  float* yb = y + (size_t)b * per_batch;
  const float n    = (float)per_batch;
  const float mean = sums[b * 2 + 0] / n;
  const float var  = sums[b * 2 + 1] / n - mean * mean;
  const float inv  = rsqrtf(var + EPSF);
  for (int i = blockIdx.x * blockDim.x + threadIdx.x; i < per_batch;
       i += blockDim.x * gridDim.x) {
    yb[i] = (xb[i] - mean) * inv * g[i] + bb[i];
  }
}

// ---------------------------------------------------------------------------
// Launch: QKV gemms -> attention(+resid) -> LN1 -> FF1(relu) -> FF2(+resid) -> LN2
// ---------------------------------------------------------------------------
extern "C" void kernel_launch(void* const* d_in, const int* in_sizes, int n_in,
                              void* d_out, int out_size, void* d_ws, size_t ws_size,
                              hipStream_t stream)
{
  const float* x      = (const float*)d_in[0];
  const float* wq     = (const float*)d_in[1];
  const float* bq     = (const float*)d_in[2];
  const float* wk     = (const float*)d_in[3];
  const float* bk     = (const float*)d_in[4];
  const float* wv     = (const float*)d_in[5];
  const float* bv     = (const float*)d_in[6];
  const float* wf1    = (const float*)d_in[7];
  const float* bf1    = (const float*)d_in[8];
  const float* wf2    = (const float*)d_in[9];
  const float* bf2    = (const float*)d_in[10];
  const float* ln_g   = (const float*)d_in[11];
  const float* ln_b   = (const float*)d_in[12];
  const float* ffln_g = (const float*)d_in[13];
  const float* ffln_b = (const float*)d_in[14];

  float* ws = (float*)d_ws;
  const size_t SLAB = (size_t)MROWS * UDIM;   // 4096*1024 floats
  float* qb   = ws;
  float* kb   = ws + 1 * SLAB;
  float* vb   = ws + 2 * SLAB;
  float* res1 = ws + 3 * SLAB;
  float* ln1  = ws + 4 * SLAB;
  float* ff2  = ws + 5 * SLAB;
  float* ff1  = ws;                           // reuses q/k/v/res1 (dead by then)
  float* sums = ws + 6 * SLAB;                // 8 floats

  const dim3 blk(256);
  const int PER_B = SEQ * UDIM;

  // QKV projections: M=4096, K=1024, N=1024 -> (4096/128)*(1024/64)=512 blocks
  wmma_gemm_kernel<<<512, blk, 0, stream>>>(x, wq, bq, nullptr, qb, MROWS, UDIM, UDIM, 0);
  wmma_gemm_kernel<<<512, blk, 0, stream>>>(x, wk, bk, nullptr, kb, MROWS, UDIM, UDIM, 0);
  wmma_gemm_kernel<<<512, blk, 0, stream>>>(x, wv, bv, nullptr, vb, MROWS, UDIM, UDIM, 0);

  // attention + residual: B*H*(S/16)=4096 waves / 8 per block = 512 blocks
  attn_kernel<<<512, blk, 0, stream>>>(qb, kb, vb, x, res1);

  // LayerNorm 1
  zero_kernel<<<1, 8, 0, stream>>>(sums, 8);
  ln_reduce_kernel<<<dim3(256, BATCH), blk, 0, stream>>>(res1, sums, PER_B);
  ln_apply_kernel<<<dim3(512, BATCH), blk, 0, stream>>>(res1, sums, ln_g, ln_b, ln1, PER_B);

  // FFN
  wmma_gemm_kernel<<<(MROWS / 128) * (FFH / 64), blk, 0, stream>>>(
      ln1, wf1, bf1, nullptr, ff1, MROWS, UDIM, FFH, 1);
  wmma_gemm_kernel<<<512, blk, 0, stream>>>(
      ff1, wf2, bf2, ln1 /*residual*/, ff2, MROWS, FFH, UDIM, 0);

  // LayerNorm 2 -> d_out
  ln_reduce_kernel<<<dim3(256, BATCH), blk, 0, stream>>>(ff2, sums + 4, PER_B);
  ln_apply_kernel<<<dim3(512, BATCH), blk, 0, stream>>>(
      ff2, sums + 4, ffln_g, ffln_b, (float*)d_out, PER_B);
}
